// DEC_LargeCNN2Int_74929999446971
// MI455X (gfx1250) — matrix operand order repaired and checked by reading, not compiled, not run on hardware
//
#include <hip/hip_runtime.h>
#include <cstddef>
#include <cstdint>

typedef __attribute__((ext_vector_type(16))) _Float16 v16h;
typedef __attribute__((ext_vector_type(8)))  _Float16 v8h;
typedef __attribute__((ext_vector_type(4)))  _Float16 v4h;
typedef __attribute__((ext_vector_type(8)))  float    v8f;

constexpr int B_     = 256;
constexpr int L_     = 2048;
constexpr int UNIT   = 100;
constexpr int FT     = 5;
constexpr int KW     = 5;
constexpr int NTILES = 7;           // 112 output channels padded (7 x 16)
constexpr int TILE_L = 256;         // l positions per workgroup
constexpr int LBLK   = L_ / TILE_L; // 8
constexpr int NP     = B_ * L_;     // 524288

// ---------------------------------------------------------------- utilities
__global__ void fill_zero_f32(float* p, int n) {
  int i = blockIdx.x * blockDim.x + threadIdx.x;
  if (i < n) p[i] = 0.0f;
}

__global__ void invperm_k(const int* __restrict__ p, int* __restrict__ inv, int n) {
  int i = blockIdx.x * blockDim.x + threadIdx.x;
  if (i < n) inv[p[i]] = i;
}

// ---------------- one-time weight prep: padded, transposed, f16 ----------------
// main convs: 48 layers of Bp[o][kk], o in [0,112), kk in [0,512); kk = k*100 + c
__global__ void prep_w_main(const float* __restrict__ w, _Float16* __restrict__ bp) {
  int i = blockIdx.x * blockDim.x + threadIdx.x;
  if (i >= 48 * 112 * 512) return;
  int lay = i / (112 * 512);
  int rem = i - lay * (112 * 512);
  int o = rem >> 9;
  int kk = rem & 511;
  int k = kk / 100, c = kk - k * 100;
  float v = 0.0f;
  if (o < UNIT && k < KW)
    v = w[(size_t)lay * UNIT * UNIT * KW + ((size_t)o * UNIT + c) * KW + k];
  bp[i] = (_Float16)v;
}

// conv0: 12 layers of Bp[o][kk], kk in [0,64); kk = k*8 + c (channel padded 7->8)
__global__ void prep_w0(const float* __restrict__ w, _Float16* __restrict__ bp) {
  int i = blockIdx.x * blockDim.x + threadIdx.x;
  if (i >= 12 * 112 * 64) return;
  int lay = i / (112 * 64);
  int rem = i - lay * (112 * 64);
  int o = rem >> 6;
  int kk = rem & 63;
  int k = kk >> 3, c = kk & 7;
  float v = 0.0f;
  if (o < UNIT && k < KW && c < 7)
    v = w[(size_t)lay * UNIT * 7 * KW + ((size_t)o * 7 + c) * KW + k];
  bp[i] = (_Float16)v;
}

// build CNN1 input: [sys interleaved(p1), par1, prior interleaved(p1), zero-pad] -> 8ch f16
__global__ void pack1_k(const float* __restrict__ recv, const int* __restrict__ p1,
                        const float* __restrict__ prior, _Float16* __restrict__ xin) {
  int t = blockIdx.x * blockDim.x + threadIdx.x;
  if (t >= NP) return;
  int b = t / L_, l = t - b * L_;
  int pl = p1[l];
  _Float16* o = xin + (size_t)t * 8;
  o[0] = (_Float16)recv[((size_t)b * L_ + pl) * 3 + 0];
  o[1] = (_Float16)recv[((size_t)b * L_ + l) * 3 + 1];
  const float* pr = prior + ((size_t)b * L_ + pl) * FT;
#pragma unroll
  for (int f = 0; f < FT; ++f) o[2 + f] = (_Float16)pr[f];
  o[7] = (_Float16)0.0f;
}

// build CNN2 input: [sys interleaved(p2), par2, il2(de1(x_plr)), zero-pad]; also save x_plr_int
__global__ void pack2_k(const float* __restrict__ recv, const int* __restrict__ p2,
                        const int* __restrict__ inv1, const float* __restrict__ xplr,
                        _Float16* __restrict__ xin, float* __restrict__ xint) {
  int t = blockIdx.x * blockDim.x + threadIdx.x;
  if (t >= NP) return;
  int b = t / L_, l = t - b * L_;
  int pl2 = p2[l];
  int j = inv1[pl2];
  _Float16* o = xin + (size_t)t * 8;
  o[0] = (_Float16)recv[((size_t)b * L_ + pl2) * 3 + 0];
  o[1] = (_Float16)recv[((size_t)b * L_ + l) * 3 + 2];
  const float* v = xplr + ((size_t)b * L_ + j) * FT;
#pragma unroll
  for (int f = 0; f < FT; ++f) {
    float vf = v[f];
    o[2 + f] = (_Float16)vf;
    xint[(size_t)t * FT + f] = vf;
  }
  o[7] = (_Float16)0.0f;
}

// ------------------------------------------------------- WMMA conv1d (SAME, K=5) + bias + ELU
// conv as GEMM: A row (output pos l) = contiguous X[b, l-2 .. l+2, 0..CINPAD) (flat kk),
// weights pre-transposed/padded f16: wp[o*PADK + kk].
template <int CINPAD, int PADK, int CHUNKK>
__global__ __launch_bounds__(256) void conv_wmma_k(const _Float16* __restrict__ xin,
                                                   const _Float16* __restrict__ wp,
                                                   const float* __restrict__ bias,
                                                   _Float16* __restrict__ xout) {
  constexpr int R      = 256 + (PADK - 1) / CINPAD;  // staged rows (halo + K-padding slack)
  constexpr int NSTEP  = CHUNKK / 32;
  constexpr int NCHUNK = PADK / CHUNKK;
  constexpr int SEGS   = CHUNKK / 8;                 // v8h segments per weight row

  __shared__ __align__(16) _Float16 Xs[R * CINPAD];
  __shared__ __align__(16) _Float16 Bls[NTILES * 16 * CHUNKK];

  const int tid = threadIdx.x;
  const int b   = blockIdx.x / LBLK;
  const int l0  = (blockIdx.x - b * LBLK) * TILE_L;

  // ---- stage activation slice X[b, l0-2 .. l0-2+R) into LDS
  const _Float16* xb = xin + (size_t)b * L_ * CINPAD;
  if (l0 >= 2 && (l0 - 2 + R) <= L_) {
    // interior: one contiguous 16B-aligned region -> vector copy
    const _Float16* src = xb + (size_t)(l0 - 2) * CINPAD;
    constexpr int NV8 = (R * CINPAD) / 8;
    for (int i = tid; i < NV8; i += 256)
      *(v8h*)&Xs[i * 8] = *(const v8h*)&src[(size_t)i * 8];
    constexpr int TAIL = R * CINPAD - NV8 * 8;
    if (TAIL > 0 && tid < TAIL) Xs[NV8 * 8 + tid] = src[(size_t)NV8 * 8 + tid];
  } else {
    // boundary: per-element with zero fill outside [0,L)
    for (int i = tid; i < R * CINPAD; i += 256) {
      int r = i / CINPAD, c = i - r * CINPAD;
      int gl = l0 - 2 + r;
      _Float16 v = (_Float16)0.0f;
      if (gl >= 0 && gl < L_) v = xb[(size_t)gl * CINPAD + c];
      Xs[i] = v;
    }
  }
  __syncthreads();

  const int lane = tid & 31;
  const int wv   = tid >> 5;   // wave id 0..7 -> rows [wv*32, wv*32+32)
  const int mrow = lane & 15;  // A: M row ; B/C/D: N column
  const int half = lane >> 4;

  v8f acc[2][NTILES];
  v8f zero = {0.f, 0.f, 0.f, 0.f, 0.f, 0.f, 0.f, 0.f};
#pragma unroll
  for (int mt = 0; mt < 2; ++mt)
#pragma unroll
    for (int nt = 0; nt < NTILES; ++nt) acc[mt][nt] = zero;

  union AF { v16h v; v4h q[4]; };
  union BF { v16h v; v8h h2[2]; };

  for (int kc = 0; kc < NCHUNK; ++kc) {
    // ---- stage weight chunk: contiguous b128 copies from prepped f16 weights
    for (int i = tid; i < NTILES * 16 * SEGS; i += 256) {
      int o = i / SEGS, sg = i - o * SEGS;
      *(v8h*)&Bls[o * CHUNKK + sg * 8] =
          *(const v8h*)&wp[(size_t)o * PADK + kc * CHUNKK + sg * 8];
    }
    __syncthreads();

#pragma unroll
    for (int s = 0; s < NSTEP; ++s) {
      const int kkA = kc * CHUNKK + s * 32 + half * 8;  // A frag: K kkA..+7 and kkA+16..+23
      AF a0, a1;
      {
        int base0 = (wv * 32 + mrow) * CINPAD + kkA;
        a0.q[0] = *(const v4h*)&Xs[base0];
        a0.q[1] = *(const v4h*)&Xs[base0 + 4];
        a0.q[2] = *(const v4h*)&Xs[base0 + 16];
        a0.q[3] = *(const v4h*)&Xs[base0 + 20];
        int base1 = (wv * 32 + 16 + mrow) * CINPAD + kkA;
        a1.q[0] = *(const v4h*)&Xs[base1];
        a1.q[1] = *(const v4h*)&Xs[base1 + 4];
        a1.q[2] = *(const v4h*)&Xs[base1 + 16];
        a1.q[3] = *(const v4h*)&Xs[base1 + 20];
      }
#pragma unroll
      for (int nt = 0; nt < NTILES; ++nt) {
        BF bf;  // B frag: column n = nt*16+mrow, K (s*32 + half*8) .. +7 and +16..+23
        int bb = (nt * 16 + mrow) * CHUNKK + s * 32 + half * 8;
        bf.h2[0] = *(const v8h*)&Bls[bb];
        bf.h2[1] = *(const v8h*)&Bls[bb + 16];
        acc[0][nt] = __builtin_amdgcn_wmma_f32_16x16x32_f16(
            false, a0.v, false, bf.v, (short)0, acc[0][nt], false, false);
        acc[1][nt] = __builtin_amdgcn_wmma_f32_16x16x32_f16(
            false, a1.v, false, bf.v, (short)0, acc[1][nt], false, false);
      }
    }
    __syncthreads();
  }

  // ---- epilogue: bias + ELU, store f16 [b][l][100]
  _Float16* yb = xout + (size_t)b * L_ * UNIT;
#pragma unroll
  for (int nt = 0; nt < NTILES; ++nt) {
    int o = nt * 16 + mrow;  // C/D: N = lane%16
    if (o < UNIT) {
      float bv = bias[o];
#pragma unroll
      for (int mt = 0; mt < 2; ++mt) {
#pragma unroll
        for (int v = 0; v < 8; ++v) {
          int rl = wv * 32 + mt * 16 + v + half * 8;  // C/D: M = vgpr + 8*(lane>=16)
          float y = acc[mt][nt][v] + bv;
          y = (y > 0.0f) ? y : (expf(y) - 1.0f);
          yb[(size_t)(l0 + rl) * UNIT + o] = (_Float16)y;
        }
      }
    }
  }
}

// ------------------------------------------------------------------ linear heads
__global__ void lin1_k(const _Float16* __restrict__ x, const float* __restrict__ w,
                       const float* __restrict__ bw, const float* __restrict__ prior,
                       const int* __restrict__ p1, float* __restrict__ xplr) {
  int t = blockIdx.x * blockDim.x + threadIdx.x;
  if (t >= NP) return;
  int b = t / L_, l = t - b * L_;
  const _Float16* xp = x + (size_t)t * UNIT;
  float acc[FT];
#pragma unroll
  for (int f = 0; f < FT; ++f) acc[f] = bw[f];
  for (int u = 0; u < UNIT; ++u) {
    float xv = (float)xp[u];
#pragma unroll
    for (int f = 0; f < FT; ++f) acc[f] += xv * w[f * UNIT + u];
  }
  int pl = p1[l];
  const float* pr = prior + ((size_t)b * L_ + pl) * FT;
#pragma unroll
  for (int f = 0; f < FT; ++f) xplr[(size_t)t * FT + f] = acc[f] - pr[f];  // extrinsic
}

__global__ void lin2_k(const _Float16* __restrict__ x, const float* __restrict__ w,
                       const float* __restrict__ bw, const float* __restrict__ xint,
                       const int* __restrict__ p2, float* __restrict__ prior) {
  int t = blockIdx.x * blockDim.x + threadIdx.x;
  if (t >= NP) return;
  int b = t / L_, l = t - b * L_;
  const _Float16* xp = x + (size_t)t * UNIT;
  float acc[FT];
#pragma unroll
  for (int f = 0; f < FT; ++f) acc[f] = bw[f];
  for (int u = 0; u < UNIT; ++u) {
    float xv = (float)xp[u];
#pragma unroll
    for (int f = 0; f < FT; ++f) acc[f] += xv * w[f * UNIT + u];
  }
  // prior = de2(x_plr - x_plr_int): scatter prior[b, p2[l]] = value at l
  float* pr = prior + ((size_t)b * L_ + p2[l]) * FT;
#pragma unroll
  for (int f = 0; f < FT; ++f) pr[f] = acc[f] - xint[(size_t)t * FT + f];
}

__global__ void lin2last_k(const _Float16* __restrict__ x, const float* __restrict__ w,
                           const float* __restrict__ bw, const int* __restrict__ p2,
                           float* __restrict__ out) {
  int t = blockIdx.x * blockDim.x + threadIdx.x;
  if (t >= NP) return;
  int b = t / L_, l = t - b * L_;
  const _Float16* xp = x + (size_t)t * UNIT;
  float s = bw[0];
  for (int u = 0; u < UNIT; ++u) s += (float)xp[u] * w[u];
  out[(size_t)b * L_ + p2[l]] = 1.0f / (1.0f + expf(-s));  // sigmoid(de2(.))
}

// ------------------------------------------------------------------ driver
extern "C" void kernel_launch(void* const* d_in, const int* in_sizes, int n_in,
                              void* d_out, int out_size, void* d_ws, size_t ws_size,
                              hipStream_t stream) {
  const float* recv    = (const float*)d_in[0];
  const int*   p1      = (const int*)d_in[1];
  const int*   p2      = (const int*)d_in[2];
  const float* conv0_w = (const float*)d_in[3];
  const float* conv0_b = (const float*)d_in[4];
  const float* convs_w = (const float*)d_in[5];
  const float* convs_b = (const float*)d_in[6];
  const float* lin1_w  = (const float*)d_in[7];
  const float* lin1_b  = (const float*)d_in[8];
  const float* lin2_w  = (const float*)d_in[9];
  const float* lin2_b  = (const float*)d_in[10];
  const float* lin2lw  = (const float*)d_in[11];
  const float* lin2lb  = (const float*)d_in[12];
  float* out = (float*)d_out;

  char* ws = (char*)d_ws;
  size_t off = 0;
  auto carve = [&](size_t bytes) -> char* {
    char* p = ws + off;
    off = (off + bytes + 255) & ~(size_t)255;
    return p;
  };
  float*    prior = (float*)carve((size_t)NP * FT * 4);
  float*    xplr  = (float*)carve((size_t)NP * FT * 4);
  float*    xint  = (float*)carve((size_t)NP * FT * 4);
  int*      inv1  = (int*)carve((size_t)L_ * 4);
  _Float16* Xin   = (_Float16*)carve((size_t)NP * 8 * 2);
  _Float16* XA    = (_Float16*)carve((size_t)NP * UNIT * 2);
  _Float16* XB    = (_Float16*)carve((size_t)NP * UNIT * 2);
  _Float16* Bp0   = (_Float16*)carve((size_t)12 * 112 * 64 * 2);
  _Float16* Bp1   = (_Float16*)carve((size_t)48 * 112 * 512 * 2);

  const int TPB  = 256;
  const int gPts  = (NP + TPB - 1) / TPB;
  const int gConv = B_ * LBLK;

  fill_zero_f32<<<(NP * FT + TPB - 1) / TPB, TPB, 0, stream>>>(prior, NP * FT);
  invperm_k<<<(L_ + TPB - 1) / TPB, TPB, 0, stream>>>(p1, inv1, L_);
  prep_w0<<<(12 * 112 * 64 + TPB - 1) / TPB, TPB, 0, stream>>>(conv0_w, Bp0);
  prep_w_main<<<(48 * 112 * 512 + TPB - 1) / TPB, TPB, 0, stream>>>(convs_w, Bp1);

  for (int idx = 0; idx < 6; ++idx) {
    for (int br = 0; br < 2; ++br) {
      if (br == 0) {
        pack1_k<<<gPts, TPB, 0, stream>>>(recv, p1, prior, Xin);
      } else {
        pack2_k<<<gPts, TPB, 0, stream>>>(recv, p2, inv1, xplr, Xin, xint);
      }
      int lay0 = idx * 2 + br;
      conv_wmma_k<8, 64, 64><<<gConv, TPB, 0, stream>>>(
          Xin, Bp0 + (size_t)lay0 * 112 * 64, conv0_b + (size_t)lay0 * UNIT, XA);
      _Float16 *src = XA, *dst = XB;
      for (int ly = 0; ly < 4; ++ly) {
        int lay = lay0 * 4 + ly;
        conv_wmma_k<100, 512, 128><<<gConv, TPB, 0, stream>>>(
            src, Bp1 + (size_t)lay * 112 * 512, convs_b + (size_t)lay * UNIT, dst);
        _Float16* tmp = src; src = dst; dst = tmp;
      }
      // after 4 swaps the final activation is in XA
      if (br == 0) {
        lin1_k<<<gPts, TPB, 0, stream>>>(XA, lin1_w + (size_t)idx * FT * UNIT,
                                         lin1_b + (size_t)idx * FT, prior, p1, xplr);
      } else if (idx < 5) {
        lin2_k<<<gPts, TPB, 0, stream>>>(XA, lin2_w + (size_t)idx * FT * UNIT,
                                         lin2_b + (size_t)idx * FT, xint, p2, prior);
      } else {
        lin2last_k<<<gPts, TPB, 0, stream>>>(XA, lin2lw, lin2lb, p2, out);
      }
    }
  }
}
